// EdgeRegressionGNN_6863357739593
// MI455X (gfx1250) — matrix operand order repaired and checked by reading, not compiled, not run on hardware
//
#include <hip/hip_runtime.h>
#include <hip/hip_bf16.h>

// ---------------- problem constants (match reference) ----------------
#define NN      100000
#define EE      1600000
#define HID     128
#define NODE_IN 32
#define EDGE_IN 16
#define EMB     32
#define FEAT    64          // EMB + NODE_IN
#define KPAD    288         // 2*HID + EDGE_IN padded to multiple of 32
#define EPS     1e-5f

#if __has_builtin(__builtin_amdgcn_global_load_async_to_lds_b128) && \
    __has_builtin(__builtin_amdgcn_s_wait_asynccnt)
#define USE_ASYNC_LDS 1
#else
#define USE_ASYNC_LDS 0
#endif

typedef float  v2f  __attribute__((ext_vector_type(2)));
typedef float  v8f  __attribute__((ext_vector_type(8)));
typedef __bf16 v16bf __attribute__((ext_vector_type(16)));

#if USE_ASYNC_LDS
// async-to-LDS builtin parameter types (from clang diagnostic):
//   arg0: int __attribute__((vector_size(16))) __device__ *   (global int4*)
//   arg1: LDS-side int4* (AS3)
typedef int v4i_async __attribute__((vector_size(16)));
typedef v4i_async __attribute__((address_space(1)))* gv4p;
typedef v4i_async __attribute__((address_space(3)))* sv4p;
#endif

union Frag16 { uint4 u[2]; v16bf v; };

// f32 WMMA 16x16x4 helper (full fp32 precision for node-side GEMMs)
__device__ __forceinline__ v8f wmma_f32x4(v2f a, v2f b, v8f c) {
    return __builtin_amdgcn_wmma_f32_16x16x4_f32(false, a, false, b, (short)0, c, false, false);
}
// bf16 WMMA 16x16x32 helper (edge MLP, f32 accumulate)
__device__ __forceinline__ v8f wmma_bf16(v16bf a, v16bf b, v8f c) {
    return __builtin_amdgcn_wmma_f32_16x16x32_bf16(false, a, false, b, (short)0, c, false, false);
}

// ---------------------------------------------------------------------
// 1) NAICS embedding gather + LayerNorm; pack feat = [LN(emb) | x]  [N,64]
//    one wave (32 lanes) per node; EMB==32 matches wave32 exactly.
// ---------------------------------------------------------------------
__global__ void k_embed_ln(const float* __restrict__ emb_tab, const int* __restrict__ idx,
                           const float* __restrict__ g, const float* __restrict__ b,
                           const float* __restrict__ x, float* __restrict__ feat) {
    int lane = threadIdx.x & 31;
    int node = blockIdx.x * 8 + (threadIdx.x >> 5);
    float v = emb_tab[(size_t)idx[node] * EMB + lane];
    float s = v, s2 = v * v;
    #pragma unroll
    for (int o = 16; o > 0; o >>= 1) { s += __shfl_xor(s, o); s2 += __shfl_xor(s2, o); }
    float mu  = s * (1.0f / 32.0f);
    float var = s2 * (1.0f / 32.0f) - mu * mu;
    float nv  = (v - mu) * rsqrtf(var + EPS) * g[lane] + b[lane];
    feat[(size_t)node * FEAT + lane]      = nv;
    feat[(size_t)node * FEAT + 32 + lane] = x[(size_t)node * NODE_IN + lane];
}

// ---------------------------------------------------------------------
// 2) degree count at dst (col)
// ---------------------------------------------------------------------
__global__ void k_deg(const int* __restrict__ col, float* __restrict__ deg) {
    int e = blockIdx.x * 256 + threadIdx.x;
    if (e < EE) atomicAdd(&deg[col[e]], 1.0f);
}

// ---------------------------------------------------------------------
// 3) weight pre-swizzle into WMMA bf16 B-fragment order:
//    frag(kb, n, half) = 16 contiguous bf16, elem e -> K = kb*32 + half*16 + e
//    Bswz1: We1 padded [288,128], Bswz2: We2 [128,64]
// ---------------------------------------------------------------------
__global__ void k_prep_w(const float* __restrict__ We1, const float* __restrict__ We2,
                         __bf16* __restrict__ Bswz1, __bf16* __restrict__ Bswz2) {
    int o = blockIdx.x * 256 + threadIdx.x;
    if (o < KPAD * HID) {           // 36864
        int e = o & 15, half = (o >> 4) & 1, n = (o >> 5) & 127, kb = o >> 12;
        int k = kb * 32 + half * 16 + e;
        Bswz1[o] = (__bf16)((k < 2 * HID + EDGE_IN) ? We1[(size_t)k * HID + n] : 0.0f);
    } else if (o < KPAD * HID + HID * 64) {   // + 8192
        int o2 = o - KPAD * HID;
        int e = o2 & 15, half = (o2 >> 4) & 1, n = (o2 >> 5) & 63, kb = o2 >> 11;
        int k = kb * 32 + half * 16 + e;
        Bswz2[o2] = (__bf16)We2[(size_t)k * 64 + n];
    }
}

// ---------------------------------------------------------------------
// 4) input projection: h = relu(feat @ W_in + b_in)   [N,128]
//    f32 WMMA 16x16x4; block = 16 rows, 8 waves x 16 cols
// ---------------------------------------------------------------------
__global__ void k_gemm_in(const float* __restrict__ feat, const float* __restrict__ Win,
                          const float* __restrict__ bin, float* __restrict__ h) {
    int lane = threadIdx.x & 31, w = threadIdx.x >> 5;
    int half = lane >> 4, ln = lane & 15;
    int m0 = blockIdx.x * 16, n0 = w * 16;
    const float* arow = feat + (size_t)(m0 + ln) * FEAT;
    v8f acc = {};
    #pragma unroll
    for (int k0 = 0; k0 < FEAT; k0 += 4) {
        float2 av = *(const float2*)(arow + k0 + 2 * half);
        v2f a, b;
        a[0] = av.x; a[1] = av.y;
        b[0] = Win[(size_t)(k0 + 2 * half) * HID + n0 + ln];
        b[1] = Win[(size_t)(k0 + 2 * half + 1) * HID + n0 + ln];
        acc = wmma_f32x4(a, b, acc);
    }
    float bias = bin[n0 + ln];
    #pragma unroll
    for (int v = 0; v < 8; ++v) {
        int m = m0 + half * 8 + v;
        h[(size_t)m * HID + n0 + ln] = fmaxf(acc[v] + bias, 0.0f);
    }
}

// ---------------------------------------------------------------------
// 5) scatter: agg[col] += h[row]  (float4 per thread, 4 f32 atomics)
// ---------------------------------------------------------------------
__global__ void k_scatter(const int* __restrict__ row, const int* __restrict__ col,
                          const float* __restrict__ h, float* __restrict__ agg) {
    unsigned tid = blockIdx.x * 256u + threadIdx.x;
    int e = tid >> 5, q = tid & 31;
    int r = row[e], c = col[e];
    float4 v = *(const float4*)(h + (size_t)r * HID + q * 4);
    float* dst = agg + (size_t)c * HID + q * 4;
    atomicAdd(dst + 0, v.x); atomicAdd(dst + 1, v.y);
    atomicAdd(dst + 2, v.z); atomicAdd(dst + 3, v.w);
}

// ---------------------------------------------------------------------
// 6) SAGE linear: out = (agg*inv_deg)@Wl + h@Wr + bl  (in-place into agg)
//    also accumulates per-column sum / sumsq for BatchNorm
// ---------------------------------------------------------------------
__global__ void k_sage(float* __restrict__ agg, const float* __restrict__ h,
                       const float* __restrict__ Wl, const float* __restrict__ bl,
                       const float* __restrict__ Wr, const float* __restrict__ deg,
                       float* __restrict__ bnstat) {
    int lane = threadIdx.x & 31, w = threadIdx.x >> 5;
    int half = lane >> 4, ln = lane & 15;
    int m0 = blockIdx.x * 16, n0 = w * 16;
    int mA = m0 + ln;
    float idg = 1.0f / fmaxf(deg[mA], 1.0f);
    const float* arow = agg + (size_t)mA * HID;
    const float* hrow = h   + (size_t)mA * HID;
    v8f acc = {};
    #pragma unroll 8
    for (int k0 = 0; k0 < HID; k0 += 4) {
        float2 av = *(const float2*)(arow + k0 + 2 * half);
        v2f a, b;
        a[0] = av.x * idg; a[1] = av.y * idg;
        b[0] = Wl[(size_t)(k0 + 2 * half) * HID + n0 + ln];
        b[1] = Wl[(size_t)(k0 + 2 * half + 1) * HID + n0 + ln];
        acc = wmma_f32x4(a, b, acc);
    }
    #pragma unroll 8
    for (int k0 = 0; k0 < HID; k0 += 4) {
        float2 av = *(const float2*)(hrow + k0 + 2 * half);
        v2f a, b;
        a[0] = av.x; a[1] = av.y;
        b[0] = Wr[(size_t)(k0 + 2 * half) * HID + n0 + ln];
        b[1] = Wr[(size_t)(k0 + 2 * half + 1) * HID + n0 + ln];
        acc = wmma_f32x4(a, b, acc);
    }
    __syncthreads();  // all waves done reading agg rows before in-place write
    float bias = bl[n0 + ln];
    float s = 0.0f, s2 = 0.0f;
    #pragma unroll
    for (int v = 0; v < 8; ++v) {
        int m = m0 + half * 8 + v;
        float o = acc[v] + bias;
        agg[(size_t)m * HID + n0 + ln] = o;
        s += o; s2 += o * o;
    }
    atomicAdd(&bnstat[n0 + ln], s);
    atomicAdd(&bnstat[HID + n0 + ln], s2);
}

// ---------------------------------------------------------------------
// 7) BN finalize: mean / inv_std per column
// ---------------------------------------------------------------------
__global__ void k_bn_fin(const float* __restrict__ bnstat, float* __restrict__ mi) {
    int c = threadIdx.x;
    float mean = bnstat[c] * (1.0f / (float)NN);
    float var  = bnstat[HID + c] * (1.0f / (float)NN) - mean * mean;
    mi[c] = mean;
    mi[HID + c] = rsqrtf(var + EPS);
}

// ---------------------------------------------------------------------
// 8) BN + tanh + residual (+ initial on odd layers)
// ---------------------------------------------------------------------
__global__ void k_bn_act(const float* __restrict__ agg, float* __restrict__ h,
                         const float* __restrict__ initial, const float* __restrict__ mi,
                         const float* __restrict__ g, const float* __restrict__ b,
                         int addInit) {
    unsigned tid = blockIdx.x * 256u + threadIdx.x;
    int c = tid & (HID - 1);
    float o = agg[tid];
    float val = tanhf((o - mi[c]) * mi[HID + c] * g[c] + b[c]) + h[tid];
    if (addInit) val += initial[tid];
    h[tid] = val;
}

// ---------------------------------------------------------------------
// 9) h -> bf16 for the memory-bound edge gather (halves HBM traffic)
// ---------------------------------------------------------------------
__global__ void k_h2bf16(const float* __restrict__ h, __bf16* __restrict__ h16) {
    unsigned tid = blockIdx.x * 256u + threadIdx.x;
    h16[tid] = (__bf16)h[tid];
}

// ---------------------------------------------------------------------
// 10) fused 3-layer edge MLP: 64 edges/block, 8 waves.
//     A-tile [64 x 288] bf16 in LDS (h[row] | h[col] | edge_attr | pad)
//     gather via GLOBAL_LOAD_ASYNC_TO_LDS_B128 (ASYNCcnt) when available;
//     L1: [64,288]@[288,128] bf16 wmma -> relu -> z(bf16, LDS)
//     L2: [64,128]@[128,64]  bf16 wmma -> relu -> z2(f32, LDS)
//     L3: dot with We3 + be3 -> d_out
// ---------------------------------------------------------------------
__global__ void k_edge_mlp(const int* __restrict__ row, const int* __restrict__ col,
                           const __bf16* __restrict__ h16, const float* __restrict__ eattr,
                           const __bf16* __restrict__ Bswz1, const __bf16* __restrict__ Bswz2,
                           const float* __restrict__ be1, const float* __restrict__ be2,
                           const float* __restrict__ We3, const float* __restrict__ be3,
                           float* __restrict__ out) {
    __shared__ __align__(16) unsigned char smem[64 * KPAD * 2];   // 36864 B
    __bf16* At = (__bf16*)smem;
    const int e0 = blockIdx.x * 64;

#if !USE_ASYNC_LDS
    // prefetch first cacheline of each gathered node row (global_prefetch_b8)
    if (threadIdx.x < 64) {
        int e = e0 + threadIdx.x;
        __builtin_prefetch(h16 + (size_t)row[e] * HID, 0, 0);
        __builtin_prefetch(h16 + (size_t)col[e] * HID, 0, 0);
    }
#endif

    // gather h16[row]|h16[col] (16B chunks): 64 edges x 32 chunks, 8 iters/thread
    for (int idx = threadIdx.x; idx < 64 * 32; idx += 256) {
        int el = idx >> 5, c = idx & 31;
        int e = e0 + el;
        int node = (c < 16) ? row[e] : col[e];
        const uint4* src = (const uint4*)(h16 + (size_t)node * HID) + (c & 15);
        uint4* dst = (uint4*)(At + el * KPAD + c * 8);
#if USE_ASYNC_LDS
        __builtin_amdgcn_global_load_async_to_lds_b128((gv4p)src, (sv4p)dst, 0, 0);
#else
        *dst = *src;
#endif
    }
    // edge_attr (f32 -> bf16) + zero pad to K=288 (overlaps with async DMA)
    for (int idx = threadIdx.x; idx < 64 * 32; idx += 256) {
        int el = idx >> 5, j = idx & 31;
        float v = (j < EDGE_IN) ? eattr[(size_t)(e0 + el) * EDGE_IN + j] : 0.0f;
        At[el * KPAD + 2 * HID + j] = (__bf16)v;
    }
#if USE_ASYNC_LDS
    __builtin_amdgcn_s_wait_asynccnt(0);   // async tile resident in LDS
#endif
    __syncthreads();

    int lane = threadIdx.x & 31, w = threadIdx.x >> 5;
    int half = lane >> 4, ln = lane & 15;
    int n0 = w * 16;

    // ---- layer 1: 4 M-tiles x 1 N-tile per wave, K = 9*32 ----
    v8f acc[4] = {};
    for (int kb = 0; kb < 9; ++kb) {
        Frag16 bf;
        const uint4* bp = (const uint4*)(Bswz1 + (size_t)kb * 4096 + (n0 + ln) * 32 + half * 16);
        bf.u[0] = bp[0]; bf.u[1] = bp[1];
        #pragma unroll
        for (int mt = 0; mt < 4; ++mt) {
            Frag16 af;
            const __bf16* ap = At + (mt * 16 + ln) * KPAD + kb * 32 + half * 16;
            af.u[0] = *(const uint4*)ap;
            af.u[1] = *(const uint4*)(ap + 8);
            acc[mt] = wmma_bf16(af.v, bf.v, acc[mt]);
        }
    }
    __syncthreads();                       // A-tile reads complete
    __bf16* Z = (__bf16*)smem;             // reuse: z [64 x 128] bf16
    float bias1 = be1[n0 + ln];
    #pragma unroll
    for (int mt = 0; mt < 4; ++mt)
        #pragma unroll
        for (int v = 0; v < 8; ++v) {
            int m = mt * 16 + half * 8 + v;
            Z[m * HID + n0 + ln] = (__bf16)fmaxf(acc[mt][v] + bias1, 0.0f);
        }
    __syncthreads();

    // ---- layer 2: wave w -> M-tile w/2, N-tiles {(w&1)*2, +1}, K = 4*32 ----
    int mt2 = w >> 1;
    v8f acc2[2] = {};
    for (int kb = 0; kb < 4; ++kb) {
        Frag16 af;
        const __bf16* ap = Z + (mt2 * 16 + ln) * HID + kb * 32 + half * 16;
        af.u[0] = *(const uint4*)ap;
        af.u[1] = *(const uint4*)(ap + 8);
        #pragma unroll
        for (int nt = 0; nt < 2; ++nt) {
            int n = (w & 1) * 32 + nt * 16;
            Frag16 bf;
            const uint4* bp = (const uint4*)(Bswz2 + (size_t)kb * 2048 + (n + ln) * 32 + half * 16);
            bf.u[0] = bp[0]; bf.u[1] = bp[1];
            acc2[nt] = wmma_bf16(af.v, bf.v, acc2[nt]);
        }
    }
    float* Z2 = (float*)(smem + 16384);    // z2 [64 x 64] f32 (disjoint from Z)
    #pragma unroll
    for (int nt = 0; nt < 2; ++nt) {
        int n = (w & 1) * 32 + nt * 16 + ln;
        float bias2 = be2[n];
        #pragma unroll
        for (int v = 0; v < 8; ++v) {
            int m = mt2 * 16 + half * 8 + v;
            Z2[m * 64 + n] = fmaxf(acc2[nt][v] + bias2, 0.0f);
        }
    }
    __syncthreads();

    // ---- layer 3: 4 lanes per edge dot z2 . We3 ----
    int e = threadIdx.x >> 2, part = threadIdx.x & 3;
    float s = 0.0f;
    #pragma unroll
    for (int j = 0; j < 16; ++j)
        s += Z2[e * 64 + part * 16 + j] * We3[part * 16 + j];
    s += __shfl_xor(s, 1);
    s += __shfl_xor(s, 2);
    if (part == 0) out[e0 + e] = s + be3[0];
}

// ---------------------------------------------------------------------
// host: pipeline launch (graph-capture safe: only async ops on stream)
// ---------------------------------------------------------------------
extern "C" void kernel_launch(void* const* d_in, const int* in_sizes, int n_in,
                              void* d_out, int out_size, void* d_ws, size_t ws_size,
                              hipStream_t stream) {
    const float* x       = (const float*)d_in[0];
    const float* eattr   = (const float*)d_in[1];
    const float* emb_tab = (const float*)d_in[2];
    const float* ln_g    = (const float*)d_in[3];
    const float* ln_b    = (const float*)d_in[4];
    const float* W_in    = (const float*)d_in[5];
    const float* b_in    = (const float*)d_in[6];
    const float* Wl      = (const float*)d_in[7];
    const float* bl      = (const float*)d_in[8];
    const float* Wr      = (const float*)d_in[9];
    const float* bn_g    = (const float*)d_in[10];
    const float* bn_b    = (const float*)d_in[11];
    const float* We1     = (const float*)d_in[12];
    const float* be1     = (const float*)d_in[13];
    const float* We2     = (const float*)d_in[14];
    const float* be2     = (const float*)d_in[15];
    const float* We3     = (const float*)d_in[16];
    const float* be3     = (const float*)d_in[17];
    const int*   ei      = (const int*)d_in[18];
    const int*   naics   = (const int*)d_in[19];
    const int* row = ei;            // src
    const int* col = ei + EE;       // dst
    float* out = (float*)d_out;

    // ---- carve workspace ----
    char* p = (char*)d_ws;
    auto carve = [&](size_t bytes) { char* r = p; p += (bytes + 255) & ~(size_t)255; return r; };
    float*  feat    = (float*)carve((size_t)NN * FEAT * 4);
    float*  h       = (float*)carve((size_t)NN * HID * 4);
    float*  initial = (float*)carve((size_t)NN * HID * 4);
    float*  agg     = (float*)carve((size_t)NN * HID * 4);
    float*  deg     = (float*)carve((size_t)NN * 4);
    float*  bnstat  = (float*)carve(2 * HID * 4);
    float*  bnmi    = (float*)carve(2 * HID * 4);
    __bf16* h16     = (__bf16*)carve((size_t)NN * HID * 2);
    __bf16* Bswz1   = (__bf16*)carve((size_t)KPAD * HID * 2);
    __bf16* Bswz2   = (__bf16*)carve((size_t)HID * 64 * 2);

    (void)hipMemsetAsync(deg, 0, (size_t)NN * 4, stream);

    k_embed_ln<<<NN / 8, 256, 0, stream>>>(emb_tab, naics, ln_g, ln_b, x, feat);
    k_deg<<<(EE + 255) / 256, 256, 0, stream>>>(col, deg);
    k_prep_w<<<(KPAD * HID + HID * 64 + 255) / 256, 256, 0, stream>>>(We1, We2, Bswz1, Bswz2);
    k_gemm_in<<<NN / 16, 256, 0, stream>>>(feat, W_in, b_in, h);
    (void)hipMemcpyAsync(initial, h, (size_t)NN * HID * 4, hipMemcpyDeviceToDevice, stream);

    for (int i = 0; i < 3; ++i) {
        (void)hipMemsetAsync(agg, 0, (size_t)NN * HID * 4, stream);
        (void)hipMemsetAsync(bnstat, 0, 2 * HID * 4, stream);
        k_scatter<<<EE * 32 / 256, 256, 0, stream>>>(row, col, h, agg);
        k_sage<<<NN / 16, 256, 0, stream>>>(agg, h, Wl + (size_t)i * HID * HID, bl + i * HID,
                                            Wr + (size_t)i * HID * HID, deg, bnstat);
        k_bn_fin<<<1, HID, 0, stream>>>(bnstat, bnmi);
        k_bn_act<<<NN * HID / 256, 256, 0, stream>>>(agg, h, initial, bnmi,
                                                     bn_g + i * HID, bn_b + i * HID, (i & 1));
    }

    k_h2bf16<<<NN * HID / 256, 256, 0, stream>>>(h, h16);
    k_edge_mlp<<<EE / 64, 256, 0, stream>>>(row, col, h16, eattr, Bswz1, Bswz2,
                                            be1, be2, We3, be3, out);
}
